// BruteForce_65687229826130
// MI455X (gfx1250) — matrix operand order repaired and checked by reading, not compiled, not run on hardware
//
#include <hip/hip_runtime.h>

// out = exp(-M) @ x
// M: 16384 x 16384 f32, x: 16384 x 128 f32, out: 16384 x 128 f32
#define N_PTS 16384
#define D_DIM 128
#define TILE_M 64   // output rows per workgroup
#define KC     32   // K chunk (one bf16 WMMA deep)
#define A_STRIDE 56 // bf16 units per LDS row (32 data + 24 pad) = 112 B
#define B_STRIDE 56

typedef __attribute__((ext_vector_type(16))) __bf16 v16bf;
typedef __attribute__((ext_vector_type(8)))  __bf16 v8bf;
typedef __attribute__((ext_vector_type(4)))  __bf16 v4bf;
typedef __attribute__((ext_vector_type(8)))  float  v8f;
typedef __attribute__((ext_vector_type(4)))  float  float4v;

union AFrag { v8bf h[2]; v16bf v; };

// exp(-m) for 4 lanes, packed to bf16 (native cvt -> v_cvt_pk_bf16_f32)
static __device__ __forceinline__ v4bf expneg4(float4v m) {
  return (v4bf){(__bf16)__expf(-m.x), (__bf16)__expf(-m.y),
                (__bf16)__expf(-m.z), (__bf16)__expf(-m.w)};
}

__global__ __launch_bounds__(256) void expm_gemm_kernel(
    const float* __restrict__ Mm, const float* __restrict__ x,
    float* __restrict__ out) {
  // double-buffered tiles: A = exp(-M) 64x32 bf16, B = x^T 128x32 bf16
  __shared__ __align__(16) __bf16 sA[2][TILE_M * A_STRIDE];
  __shared__ __align__(16) __bf16 sB[2][D_DIM * B_STRIDE];

  const int t    = threadIdx.x;
  const int wave = t >> 5;
  const int lane = t & 31;

  const int tileRow  = blockIdx.x * TILE_M;
  const int rowBlock = (wave & 3) * 16;   // 0..48 within tile
  const int colBlock = (wave >> 2) * 64;  // 0 or 64

  // ---- staging address math ----
  // M tile: rows tileRow + (t>>3) + {0,32}, cols kc + (t&7)*4 .. +3 (float4)
  const int    mRowL = t >> 3;        // 0..31
  const int    mColL = (t & 7) * 4;   // 0,4,...,28
  const float* Mp    = Mm + (size_t)(tileRow + mRowL) * N_PTS + mColL;
  const size_t MrowJump = (size_t)32 * N_PTS;

  // x tile: col = t&127, k = kc + (t>>7)*16 + i  (i = 0..15)
  const int    xCol = t & 127;
  const int    xKb  = (t >> 7) * 16;  // 0 or 16
  const float* Xp   = x + (size_t)xKb * D_DIM + xCol;

  // ---- fragment read addresses (ISA 7.12.2 layouts) ----
  // A 16x32 bf16: lane<16 row=lane,    K {0..7} & {16..23}
  //               lane>=16 row=lane-16, K {8..15} & {24..31}
  const int aRow  = rowBlock + (lane & 15);
  const int aKoff = (lane < 16) ? 0 : 8;
  // B 32x16 bf16 per tile: lane col = lane&15; lane<16 K {0..7}&{16..23} else K {8..15}&{24..31}
  const int bKb   = (lane < 16) ? 0 : 8;

  v8f acc[4];
#pragma unroll
  for (int i = 0; i < 4; ++i)
    acc[i] = (v8f){0.f, 0.f, 0.f, 0.f, 0.f, 0.f, 0.f, 0.f};

  float4v m0, m1;
  float   xv[16];

  // ---- stage k-chunk 0 into buffer 0 ----
  m0 = *(const float4v*)(Mp);
  m1 = *(const float4v*)(Mp + MrowJump);
#pragma unroll
  for (int i = 0; i < 16; ++i) xv[i] = Xp[(size_t)i * D_DIM];
  {
    *(v4bf*)&sA[0][mRowL * A_STRIDE + mColL]        = expneg4(m0);
    *(v4bf*)&sA[0][(mRowL + 32) * A_STRIDE + mColL] = expneg4(m1);
    v8bf pb0, pb1;
#pragma unroll
    for (int i = 0; i < 8; ++i) { pb0[i] = (__bf16)xv[i]; pb1[i] = (__bf16)xv[8 + i]; }
    *(v8bf*)&sB[0][xCol * B_STRIDE + xKb]     = pb0;
    *(v8bf*)&sB[0][xCol * B_STRIDE + xKb + 8] = pb1;
  }

  // ---- steady state: 511 branch-free iterations ----
  int p = 0;
  for (int kc = 0; kc < N_PTS - KC; kc += KC) {
    __syncthreads();

    // next chunk's global loads, issued early (overlap with WMMA below)
    m0 = *(const float4v*)(Mp + (kc + KC));
    m1 = *(const float4v*)(Mp + MrowJump + (kc + KC));
#pragma unroll
    for (int i = 0; i < 16; ++i)
      xv[i] = Xp[(size_t)(kc + KC + i) * D_DIM];

    // fragment loads from buffer p
    AFrag a;
    {
      const __bf16* sAp = &sA[p][aRow * A_STRIDE];
      a.h[0] = *(const v8bf*)(sAp + aKoff);        // K aoff..aoff+7
      a.h[1] = *(const v8bf*)(sAp + aKoff + 16);   // K aoff+16..aoff+23
    }
    AFrag b[4];
#pragma unroll
    for (int tn = 0; tn < 4; ++tn) {
      const __bf16* sBp =
          &sB[p][(colBlock + tn * 16 + (lane & 15)) * B_STRIDE + bKb];
      b[tn].h[0] = *(const v8bf*)(sBp);
      b[tn].h[1] = *(const v8bf*)(sBp + 16);
    }

    // convert + store next chunk to the other buffer
    {
      const int q = p ^ 1;
      *(v4bf*)&sA[q][mRowL * A_STRIDE + mColL]        = expneg4(m0);
      *(v4bf*)&sA[q][(mRowL + 32) * A_STRIDE + mColL] = expneg4(m1);
      v8bf pb0, pb1;
#pragma unroll
      for (int i = 0; i < 8; ++i) { pb0[i] = (__bf16)xv[i]; pb1[i] = (__bf16)xv[8 + i]; }
      *(v8bf*)&sB[q][xCol * B_STRIDE + xKb]     = pb0;
      *(v8bf*)&sB[q][xCol * B_STRIDE + xKb + 8] = pb1;
    }

    // 4 WMMAs: 16 rows x 64 cols, K += 32
#pragma unroll
    for (int tn = 0; tn < 4; ++tn) {
      acc[tn] = __builtin_amdgcn_wmma_f32_16x16x32_bf16(
          false, a.v, false, b[tn].v, (short)0, acc[tn], false, false);
    }
    p ^= 1;
  }

  // ---- peeled last chunk: no staging, just consume buffer p ----
  {
    __syncthreads();
    AFrag a;
    {
      const __bf16* sAp = &sA[p][aRow * A_STRIDE];
      a.h[0] = *(const v8bf*)(sAp + aKoff);
      a.h[1] = *(const v8bf*)(sAp + aKoff + 16);
    }
#pragma unroll
    for (int tn = 0; tn < 4; ++tn) {
      AFrag b;
      const __bf16* sBp =
          &sB[p][(colBlock + tn * 16 + (lane & 15)) * B_STRIDE + bKb];
      b.h[0] = *(const v8bf*)(sBp);
      b.h[1] = *(const v8bf*)(sBp + 16);
      acc[tn] = __builtin_amdgcn_wmma_f32_16x16x32_bf16(
          false, a.v, false, b.v, (short)0, acc[tn], false, false);
    }
  }

  // ---- epilogue: C layout -> global (VGPR j: M=j (lanes 0-15) / M=j+8) ----
  const int outColBase = colBlock + (lane & 15);
  const int outRowBase = tileRow + rowBlock + ((lane >> 4) << 3);
#pragma unroll
  for (int tn = 0; tn < 4; ++tn) {
#pragma unroll
    for (int j = 0; j < 8; ++j) {
      out[(size_t)(outRowBase + j) * D_DIM + outColBase + tn * 16] = acc[tn][j];
    }
  }
}

extern "C" void kernel_launch(void* const* d_in, const int* in_sizes, int n_in,
                              void* d_out, int out_size, void* d_ws, size_t ws_size,
                              hipStream_t stream) {
  const float* Mm  = (const float*)d_in[0];
  const float* x   = (const float*)d_in[1];
  float*       out = (float*)d_out;
  (void)in_sizes; (void)n_in; (void)out_size; (void)d_ws; (void)ws_size;

  dim3 grid(N_PTS / TILE_M);  // 256 workgroups
  dim3 block(256);            // 8 wave32
  expm_gemm_kernel<<<grid, block, 0, stream>>>(Mm, x, out);
}
